// Model_8117488189610
// MI455X (gfx1250) — matrix operand order repaired and checked by reading, not compiled, not run on hardware
//
#include <hip/hip_runtime.h>
#include <stdint.h>

// ---------------------------------------------------------------------------
// Types for CDNA5 WMMA
// ---------------------------------------------------------------------------
typedef __attribute__((ext_vector_type(16))) __bf16 v16bf;
typedef __attribute__((ext_vector_type(8)))  float  v8f;

union FragBF {
    v16bf          v;
    unsigned int   u32[8];
    unsigned short u16[16];
    uint4          q[2];
};

// Pack two f32 into a bf16 pair (lo -> [15:0], hi -> [31:16]) with one v_perm.
__device__ __forceinline__ unsigned int pack2_bf16(float lo, float hi) {
    union { float f; unsigned int u; } a, b;
    a.f = lo; b.f = hi;
    // round-half-up then take top 16 bits of each via byte-perm
    return __builtin_amdgcn_perm(b.u + 0x8000u, a.u + 0x8000u, 0x07060302u);
}

// Scalar f32 -> bf16 (round to nearest even), for non-critical producers.
__device__ __forceinline__ unsigned short f2bf(float f) {
    union { float f; unsigned int u; } x; x.f = f;
    unsigned int r = x.u + 0x7FFFu + ((x.u >> 16) & 1u);
    return (unsigned short)(r >> 16);
}

// ---------------------------------------------------------------------------
// Core kernel: partial C[16x32] = adj[16 rows x klen] @ Bt (bf16 WMMA)
//   adj : f32 row-major [N x N]
//   bpt : bf16, B stored TRANSPOSED: bpt[c][k], c in [0,32), k in [0,N)
//   part: f32 [KSPLIT][N][32]
// grid.x = N/16 row tiles, grid.y = KSPLIT, block = 32 (one wave)
// ---------------------------------------------------------------------------
__global__ __launch_bounds__(32)
void k_wmma_gemm(const float* __restrict__ adj,
                 const unsigned short* __restrict__ bpt,
                 float* __restrict__ part, int N, int klen) {
    const int lane = threadIdx.x;
    const int m0   = blockIdx.x * 16;
    const int s    = blockIdx.y;
    const int k0   = s * klen;
    const int lm   = lane & 15;
    const int hi   = lane >> 4;            // 0 or 1

    const float* arow = adj + (size_t)(m0 + lm) * N;
    // B columns (rows of bpt): n-tile 0 and 1; lane's K chunks {8h..8h+7, 16+8h..16+8h+7}
    const unsigned short* b0 = bpt + (size_t)(lm)      * N + hi * 8;
    const unsigned short* b1 = bpt + (size_t)(16 + lm) * N + hi * 8;

    v8f acc0 = {0.f,0.f,0.f,0.f,0.f,0.f,0.f,0.f};
    v8f acc1 = {0.f,0.f,0.f,0.f,0.f,0.f,0.f,0.f};

    for (int k = k0; k < k0 + klen; k += 32) {
        // ---- A fragment: lanes<16 take K {0..7,16..23}, lanes>=16 {8..15,24..31}
        const float* pa0 = arow + k + hi * 8;
        const float* pa1 = pa0 + 16;
        float4 f0 = *(const float4*)(pa0);
        float4 f1 = *(const float4*)(pa0 + 4);
        float4 f2 = *(const float4*)(pa1);
        float4 f3 = *(const float4*)(pa1 + 4);
        FragBF a;
        a.u32[0] = pack2_bf16(f0.x, f0.y);
        a.u32[1] = pack2_bf16(f0.z, f0.w);
        a.u32[2] = pack2_bf16(f1.x, f1.y);
        a.u32[3] = pack2_bf16(f1.z, f1.w);
        a.u32[4] = pack2_bf16(f2.x, f2.y);
        a.u32[5] = pack2_bf16(f2.z, f2.w);
        a.u32[6] = pack2_bf16(f3.x, f3.y);
        a.u32[7] = pack2_bf16(f3.z, f3.w);

        // ---- B fragments: two chunks of 8 bf16 per lane (k+8h.., k+16+8h..)
        FragBF fb0, fb1;
        fb0.q[0] = *(const uint4*)(b0 + k);
        fb0.q[1] = *(const uint4*)(b0 + k + 16);
        fb1.q[0] = *(const uint4*)(b1 + k);
        fb1.q[1] = *(const uint4*)(b1 + k + 16);

        acc0 = __builtin_amdgcn_wmma_f32_16x16x32_bf16(false, a.v, false, fb0.v,
                                                       (short)0, acc0, false, false);
        acc1 = __builtin_amdgcn_wmma_f32_16x16x32_bf16(false, a.v, false, fb1.v,
                                                       (short)0, acc1, false, false);
    }

    // ---- store partials. C layout: vgpr r -> M = r + 8*hi, N = lane&15
    float* pout = part + ((size_t)s * N + m0) * 32;
    const int mr = hi << 3;
#pragma unroll
    for (int r = 0; r < 8; ++r) {
        pout[(size_t)(mr + r) * 32 + lm]      = acc0[r];
        pout[(size_t)(mr + r) * 32 + 16 + lm] = acc1[r];
    }
}

// ---------------------------------------------------------------------------
// Small GEMM producer: bpt[c0+c][n] = bf16( sum_i in[n][i] * W[i][c] )
// ---------------------------------------------------------------------------
__global__ void k_xw(const float* __restrict__ in, const float* __restrict__ W,
                     unsigned short* __restrict__ bpt, int N, int Fin, int Fout, int c0) {
    int n = blockIdx.x * blockDim.x + threadIdx.x;
    if (n >= N) return;
    float xi[18];
    for (int i = 0; i < Fin; ++i) xi[i] = in[(size_t)n * Fin + i];
    for (int c = 0; c < Fout; ++c) {
        float s = 0.f;
        for (int i = 0; i < Fin; ++i) s += xi[i] * W[i * Fout + c];
        bpt[(size_t)(c0 + c) * N + n] = f2bf(s);
    }
}

__global__ void k_fill0(unsigned int* __restrict__ p, int n) {
    int i = blockIdx.x * blockDim.x + threadIdx.x;
    if (i < n) p[i] = 0u;
}

// ---------------------------------------------------------------------------
// Reduce KSPLIT partials + bias + optional relu -> dst[n][c] (ld = ldd)
// ---------------------------------------------------------------------------
__global__ void k_epilogue(const float* __restrict__ part, float* __restrict__ dst,
                           const float* __restrict__ bias, int N, int c0, int Fc,
                           int ldd, int relu, int ksplit) {
    int idx = blockIdx.x * blockDim.x + threadIdx.x;
    int n = idx / Fc, c = idx - n * Fc;
    if (n >= N) return;
    float s = 0.f;
    for (int t = 0; t < ksplit; ++t)
        s += part[((size_t)t * N + n) * 32 + c0 + c];
    if (bias) s += bias[c];
    if (relu) s = fmaxf(s, 0.f);
    dst[(size_t)n * ldd + c] = s;
}

// ---------------------------------------------------------------------------
// BatchNorm over N rows, one block per column. Deterministic LDS tree reduce.
// ---------------------------------------------------------------------------
__global__ __launch_bounds__(256)
void k_bn(const float* __restrict__ src, float* __restrict__ dst,
          const float* __restrict__ g, const float* __restrict__ b,
          int N, int lds_, int ldd) {
    const int col = blockIdx.x;
    const int tid = threadIdx.x;
    __shared__ float s1[256], s2[256];
    float a = 0.f, q = 0.f;
    for (int n = tid; n < N; n += 256) {
        float v = src[(size_t)n * lds_ + col];
        a += v; q += v * v;
    }
    s1[tid] = a; s2[tid] = q;
    __syncthreads();
    for (int o = 128; o > 0; o >>= 1) {
        if (tid < o) { s1[tid] += s1[tid + o]; s2[tid] += s2[tid + o]; }
        __syncthreads();
    }
    float m  = s1[0] / (float)N;
    float vv = s2[0] / (float)N - m * m;
    float sc = g[col] * rsqrtf(vv + 1e-5f);
    float sh = b[col] - m * sc;
    for (int n = tid; n < N; n += 256)
        dst[(size_t)n * ldd + col] = src[(size_t)n * lds_ + col] * sc + sh;
}

// ---------------------------------------------------------------------------
// Row softmax over 20 logits (ld 32). Writes f32 s_l and bf16 transposed+padded
// B for the next adj GEMM (rows 20..31 zeroed).
// ---------------------------------------------------------------------------
__global__ void k_softmax(const float* __restrict__ logits, float* __restrict__ slf,
                          unsigned short* __restrict__ bpt, int N) {
    int n = blockIdx.x * blockDim.x + threadIdx.x;
    if (n >= N) return;
    float v[20], mx = -3.0e38f;
    for (int c = 0; c < 20; ++c) { v[c] = logits[(size_t)n * 32 + c]; mx = fmaxf(mx, v[c]); }
    float s = 0.f;
    for (int c = 0; c < 20; ++c) { v[c] = expf(v[c] - mx); s += v[c]; }
    float inv = 1.f / s;
    for (int c = 0; c < 20; ++c) {
        float p = v[c] * inv;
        slf[(size_t)n * 20 + c] = p;
        bpt[(size_t)c * N + n]  = f2bf(p);
    }
    for (int c = 20; c < 32; ++c) bpt[(size_t)c * N + n] = 0;
}

// ---------------------------------------------------------------------------
// pool partial: pp[b][p*28+c] = sum over 128-row chunk of s_l[n][p] * M[n][c]
// where M = [ zbuf (8 cols) | AS (20 cols, ld 32) ]
// ---------------------------------------------------------------------------
__global__ __launch_bounds__(256)
void k_pool_partial(const float* __restrict__ slf, const float* __restrict__ zbuf,
                    const float* __restrict__ as, float* __restrict__ pp, int N) {
    __shared__ float S[128 * 20];
    __shared__ float M[128 * 28];
    const int r0 = blockIdx.x * 128;
    for (int i = threadIdx.x; i < 128 * 20; i += 256) {
        int r = i / 20, c = i - r * 20;
        S[i] = slf[(size_t)(r0 + r) * 20 + c];
    }
    for (int i = threadIdx.x; i < 128 * 28; i += 256) {
        int r = i / 28, c = i - r * 28;
        M[i] = (c < 8) ? zbuf[(size_t)(r0 + r) * 8 + c]
                       : as[(size_t)(r0 + r) * 32 + (c - 8)];
    }
    __syncthreads();
    for (int o = threadIdx.x; o < 560; o += 256) {
        int p = o / 28, c = o - p * 28;
        float s = 0.f;
        for (int r = 0; r < 128; ++r) s += S[r * 20 + p] * M[r * 28 + c];
        pp[(size_t)blockIdx.x * 560 + o] = s;
    }
}

__global__ void k_pool_final(const float* __restrict__ pp, float* __restrict__ hp,
                             float* __restrict__ amat, int nb) {
    int o = blockIdx.x * blockDim.x + threadIdx.x;
    if (o >= 560) return;
    float s = 0.f;
    for (int b = 0; b < nb; ++b) s += pp[(size_t)b * 560 + o];
    int p = o / 28, c = o - p * 28;
    if (c < 8) hp[p * 8 + c] = s;
    else       amat[p * 20 + (c - 8)] = s;
}

// ---------------------------------------------------------------------------
// Tail: 20x20 / 20x8 math, ~9k MACs, serial on lane 0.
// ---------------------------------------------------------------------------
__global__ void k_tail(const float* __restrict__ hp, const float* __restrict__ amat,
                       const float* __restrict__ W3, const float* __restrict__ b3,
                       const float* __restrict__ We2, const float* __restrict__ be2,
                       const float* __restrict__ g3, const float* __restrict__ bt3,
                       float* __restrict__ out) {
    if (threadIdx.x != 0 || blockIdx.x != 0) return;
    float h[20][8], a[20][20], t1[20][8], y3[20][8], h3[20][8], t2[20][8];
    for (int i = 0; i < 20; ++i)
        for (int j = 0; j < 8; ++j) h[i][j] = hp[i * 8 + j];
    for (int i = 0; i < 20; ++i)
        for (int k = 0; k < 20; ++k) a[i][k] = amat[i * 20 + k];
    // t1 = h @ W3
    for (int i = 0; i < 20; ++i)
        for (int j = 0; j < 8; ++j) {
            float s = 0.f;
            for (int k = 0; k < 8; ++k) s += h[i][k] * W3[k * 8 + j];
            t1[i][j] = s;
        }
    // y3 = relu(a @ t1 + b3)
    for (int i = 0; i < 20; ++i)
        for (int j = 0; j < 8; ++j) {
            float s = b3[j];
            for (int k = 0; k < 20; ++k) s += a[i][k] * t1[k][j];
            y3[i][j] = fmaxf(s, 0.f);
        }
    // bn over 20 rows
    for (int j = 0; j < 8; ++j) {
        float m = 0.f, q = 0.f;
        for (int i = 0; i < 20; ++i) { m += y3[i][j]; q += y3[i][j] * y3[i][j]; }
        m /= 20.f; q = q / 20.f - m * m;
        float sc = g3[j] * rsqrtf(q + 1e-5f);
        float sh = bt3[j] - m * sc;
        for (int i = 0; i < 20; ++i) h3[i][j] = y3[i][j] * sc + sh;
    }
    // t2 = h3 @ We2
    for (int i = 0; i < 20; ++i)
        for (int j = 0; j < 8; ++j) {
            float s = 0.f;
            for (int k = 0; k < 8; ++k) s += h3[i][k] * We2[k * 8 + j];
            t2[i][j] = s;
        }
    // z2 = a @ t2 + be2; out = column sum
    for (int j = 0; j < 8; ++j) {
        float o = 0.f;
        for (int i = 0; i < 20; ++i) {
            float s = be2[j];
            for (int k = 0; k < 20; ++k) s += a[i][k] * t2[k][j];
            o += s;
        }
        out[j] = o;
    }
}

// ---------------------------------------------------------------------------
extern "C" void kernel_launch(void* const* d_in, const int* in_sizes, int n_in,
                              void* d_out, int out_size, void* d_ws, size_t ws_size,
                              hipStream_t stream) {
    const float* x   = (const float*)d_in[0];
    const float* adj = (const float*)d_in[1];
    const float* W1  = (const float*)d_in[2];  const float* b1  = (const float*)d_in[3];
    const float* W2  = (const float*)d_in[4];  const float* b2  = (const float*)d_in[5];
    const float* We1 = (const float*)d_in[6];  const float* be1 = (const float*)d_in[7];
    const float* Wp1 = (const float*)d_in[8];  const float* bp1 = (const float*)d_in[9];
    const float* W3  = (const float*)d_in[10]; const float* b3  = (const float*)d_in[11];
    const float* We2 = (const float*)d_in[12]; const float* be2 = (const float*)d_in[13];
    const float* g1  = (const float*)d_in[14]; const float* bt1 = (const float*)d_in[15];
    const float* g2  = (const float*)d_in[16]; const float* bt2 = (const float*)d_in[17];
    const float* g3  = (const float*)d_in[18]; const float* bt3 = (const float*)d_in[19];

    const int N = in_sizes[0] / 18;           // 8192
    const int KSPLIT = 4;
    const int klen = N / KSPLIT;

    // ---- workspace carve-out (256B aligned)
    size_t off = 0;
    auto carve = [&](size_t bytes) {
        size_t o = off;
        off = (off + bytes + 255) & ~(size_t)255;
        return o;
    };
    char* ws = (char*)d_ws;
    unsigned short* bpt = (unsigned short*)(ws + carve((size_t)32 * N * 2));
    float* part = (float*)(ws + carve((size_t)KSPLIT * N * 32 * 4));
    float* y    = (float*)(ws + carve((size_t)N * 32 * 4));
    float* h1   = (float*)(ws + carve((size_t)N * 16 * 4));
    float* h2   = (float*)(ws + carve((size_t)N * 16 * 4));
    float* zbuf = (float*)(ws + carve((size_t)N * 8 * 4));
    float* slf  = (float*)(ws + carve((size_t)N * 20 * 4));
    float* pp   = (float*)(ws + carve((size_t)64 * 560 * 4));
    float* hpb  = (float*)(ws + carve(160 * 4));
    float* amat = (float*)(ws + carve(400 * 4));
    (void)ws_size; (void)n_in; (void)out_size;

    const dim3 gW(N / 16, KSPLIT), bW(32);
    const int fillN = 32 * N / 2;                     // u32 count of bpt
    const int gFill = (fillN + 255) / 256;
    const int gRows = (N + 255) / 256;

    // ---- pass A: h1 = bn(relu(adj @ (x@W1) + b1))
    k_fill0<<<gFill, 256, 0, stream>>>((unsigned int*)bpt, fillN);
    k_xw<<<gRows, 256, 0, stream>>>(x, W1, bpt, N, 18, 16, 0);
    k_wmma_gemm<<<gW, bW, 0, stream>>>(adj, bpt, part, N, klen);
    k_epilogue<<<(N * 16 + 255) / 256, 256, 0, stream>>>(part, y, b1, N, 0, 16, 32, 1, KSPLIT);
    k_bn<<<16, 256, 0, stream>>>(y, h1, g1, bt1, N, 32, 16);

    // ---- pass B: h2 = bn(relu(adj @ (h1@W2) + b2))
    k_fill0<<<gFill, 256, 0, stream>>>((unsigned int*)bpt, fillN);
    k_xw<<<gRows, 256, 0, stream>>>(h1, W2, bpt, N, 16, 16, 0);
    k_wmma_gemm<<<gW, bW, 0, stream>>>(adj, bpt, part, N, klen);
    k_epilogue<<<(N * 16 + 255) / 256, 256, 0, stream>>>(part, y, b2, N, 0, 16, 32, 1, KSPLIT);
    k_bn<<<16, 256, 0, stream>>>(y, h2, g2, bt2, N, 32, 16);

    // ---- pass C: [z | logits] = adj @ (h2 @ [We1 | Wp1]) + [be1 | bp1]
    k_fill0<<<gFill, 256, 0, stream>>>((unsigned int*)bpt, fillN);
    k_xw<<<gRows, 256, 0, stream>>>(h2, We1, bpt, N, 16, 8, 0);
    k_xw<<<gRows, 256, 0, stream>>>(h2, Wp1, bpt, N, 16, 20, 8);
    k_wmma_gemm<<<gW, bW, 0, stream>>>(adj, bpt, part, N, klen);
    k_epilogue<<<(N * 8 + 255) / 256, 256, 0, stream>>>(part, zbuf, be1, N, 0, 8, 8, 0, KSPLIT);
    k_epilogue<<<(N * 20 + 255) / 256, 256, 0, stream>>>(part, y, bp1, N, 8, 20, 32, 0, KSPLIT);

    // ---- softmax -> s_l (f32) + packed/transposed bf16 B for pass D
    k_softmax<<<gRows, 256, 0, stream>>>(y, slf, bpt, N);

    // ---- pass D: AS = adj @ s_l
    k_wmma_gemm<<<gW, bW, 0, stream>>>(adj, bpt, part, N, klen);
    k_epilogue<<<(N * 20 + 255) / 256, 256, 0, stream>>>(part, y, nullptr, N, 0, 20, 32, 0, KSPLIT);

    // ---- pooling: [h_pool | a] = s_l^T @ [z | AS]
    k_pool_partial<<<N / 128, 256, 0, stream>>>(slf, zbuf, y, pp, N);
    k_pool_final<<<(560 + 255) / 256, 256, 0, stream>>>(pp, hpb, amat, N / 128);

    // ---- tiny tail -> d_out[8]
    k_tail<<<1, 32, 0, stream>>>(hpb, amat, W3, b3, We2, be2, g3, bt3, (float*)d_out);
}
